// QuantLlamaAttention_13065290515022
// MI455X (gfx1250) — matrix operand order repaired and checked by reading, not compiled, not run on hardware
//
#include <hip/hip_runtime.h>
#include <hip/hip_bf16.h>

// ---------------------------------------------------------------------------
// Types for CDNA5 WMMA / TDM
// ---------------------------------------------------------------------------
typedef __attribute__((ext_vector_type(16))) __bf16 v16bf;
typedef __attribute__((ext_vector_type(8)))  float  v8f;
typedef __attribute__((ext_vector_type(4)))  unsigned int u32x4;
typedef __attribute__((ext_vector_type(8)))  int i32x8;
typedef __attribute__((ext_vector_type(4)))  int i32x4;
typedef __attribute__((address_space(3)))    unsigned short lds_us_t;

union Frag {
    int4  i4[2];   // 32 bytes
    v16bf v;       // 16 x bf16
};

#if __has_builtin(__builtin_amdgcn_tensor_load_to_lds) && \
    __has_builtin(__builtin_amdgcn_s_wait_tensorcnt)
#define HAS_TDM 1
#else
#define HAS_TDM 0
#endif

__device__ __forceinline__ unsigned short f2bf(float f) {
    union { float f; unsigned u; } x; x.f = f;
    unsigned r = x.u + 0x7FFFu + ((x.u >> 16) & 1u);   // round-to-nearest-even
    return (unsigned short)(r >> 16);
}

__device__ __forceinline__ float rowmax16(float v) {
    #pragma unroll
    for (int off = 1; off < 16; off <<= 1) v = fmaxf(v, __shfl_xor(v, off, 32));
    return v;
}
__device__ __forceinline__ float rowsum16(float v) {
    #pragma unroll
    for (int off = 1; off < 16; off <<= 1) v += __shfl_xor(v, off, 32);
    return v;
}

// ---------------------------------------------------------------------------
// Problem constants
// ---------------------------------------------------------------------------
#define SEQ     2048
#define HID     4096
#define NH      32
#define HD      128
#define QKV_N   (3 * HID)          // 12288

// ---------------------------------------------------------------------------
// Kernel 1: fp32 -> bf16 bulk convert (vectorized x4)
// ---------------------------------------------------------------------------
__global__ __launch_bounds__(256) void cvt_f32_bf16(const float* __restrict__ in,
                                                    unsigned short* __restrict__ out,
                                                    long long n) {
    long long i = ((long long)blockIdx.x * 256 + threadIdx.x) * 4;
    if (i + 3 < n) {
        float4 f = *(const float4*)(in + i);
        ushort4 u;
        u.x = f2bf(f.x); u.y = f2bf(f.y); u.z = f2bf(f.z); u.w = f2bf(f.w);
        *(ushort4*)(out + i) = u;
    }
}

// ---------------------------------------------------------------------------
// Kernel 2: bf16 GEMM  C[M,N] = A[M,K] * B[K,N]
//   block = 128x128, 256 threads = 8 waves (4 Mwaves x 2 Nwaves),
//   wave tile = 32x64 (2x4 of 16x16), K-step 32.
//   Double-buffered LDS; A panel DMA'd by the Tensor Data Mover (TDM) with
//   hardware LDS row padding; B panel staged transposed; B prefetched 2 ahead.
// ---------------------------------------------------------------------------
template <bool F32OUT>
__global__ __launch_bounds__(256) void gemm_bf16(const unsigned short* __restrict__ A,
                                                 const unsigned short* __restrict__ B,
                                                 void* __restrict__ Cv,
                                                 int M, int N, int K) {
    __shared__ __align__(16) unsigned short As[2][128][40];   // [m][k], pad 40 (TDM pad)
    __shared__ __align__(16) unsigned short Bs[2][128][40];   // B^T: [n][k]

    const int t    = threadIdx.x;
    const int wave = t >> 5;
    const int lane = t & 31;
    const int l16  = lane & 15;
    const int hf   = lane >> 4;
    const int wm   = wave >> 1;           // 0..3
    const int wn   = wave & 1;            // 0..1
    const int m_blk = blockIdx.y * 128;
    const int n_blk = blockIdx.x * 128;
    const int nt    = K >> 5;             // number of 32-wide K tiles

    // ---- stage B panel transposed into buffer `buf`
    auto stageB = [&](int k0, int buf) {
        int n  = t & 127;
        int kh = t >> 7;      // 0/1 -> k 0..15 / 16..31
        union { unsigned short u[16]; int4 i4[2]; } tb;
        const unsigned short* bp = B + (size_t)(k0 + kh * 16) * N + n_blk + n;
        #pragma unroll
        for (int j = 0; j < 16; ++j) tb.u[j] = bp[(size_t)j * N];
        *(int4*)&Bs[buf][n][kh * 16]     = tb.i4[0];
        *(int4*)&Bs[buf][n][kh * 16 + 8] = tb.i4[1];
    };

    // ---- issue A panel (128 x 32 bf16 tile) into buffer `buf`
#if HAS_TDM
    auto issueA = [&](int k0, int buf) {
        if (wave == 0) {
            // D# group 0: count=1 | lds_addr | global_addr | type=2
            unsigned lds_off = (unsigned)(size_t)(lds_us_t*)&As[buf][0][0];
            unsigned long long ga =
                (unsigned long long)(size_t)(A + (size_t)m_blk * K + k0);
            u32x4 g0;
            g0.x = 1u;                                    // count = 1 (valid)
            g0.y = lds_off;                               // lds_addr (bytes)
            g0.z = (unsigned)ga;                          // global_addr[31:0]
            g0.w = (unsigned)(ga >> 32) | 0x80000000u;    // ga[56:32] | type=2<<30
            // D# group 1: data_size=2B, pad_enable, pad_interval=16dw, pad=4dw,
            //             tensor_dim0=K, tensor_dim1=M, tile=32x128, stride0=K
            i32x8 g1;
            g1[0] = (int)((1u << 16) | (1u << 20) | (3u << 22) | (3u << 25));
            g1[1] = (int)(((unsigned)K & 0xFFFFu) << 16);                 // td0 lo
            g1[2] = (int)((((unsigned)K >> 16) & 0xFFFFu) |
                          (((unsigned)M & 0xFFFFu) << 16));               // td0 hi|td1 lo
            g1[3] = (int)(((((unsigned)M) >> 16) & 0xFFFFu) | (32u << 16)); // td1 hi|tile0
            g1[4] = 128;                                                  // tile1 | tile2=0
            g1[5] = (int)(unsigned)K;                                     // stride0[31:0]
            g1[6] = 0;
            g1[7] = 0;
            i32x4 z4 = {0, 0, 0, 0};
#if defined(__clang_major__) && (__clang_major__ >= 23)
            i32x8 z8 = {0, 0, 0, 0, 0, 0, 0, 0};
            __builtin_amdgcn_tensor_load_to_lds(g0, g1, z4, z4, z8, 0);
#else
            __builtin_amdgcn_tensor_load_to_lds(g0, g1, z4, z4, 0);
#endif
        }
    };
#else
    auto issueA = [&](int k0, int buf) {
        #pragma unroll
        for (int c = t; c < 512; c += 256) {
            int r = c >> 2, off = c & 3;
            *(int4*)&As[buf][r][off * 8] =
                *(const int4*)(A + (size_t)(m_blk + r) * K + k0 + off * 8);
        }
    };
#endif

    v8f acc[2][4];
    #pragma unroll
    for (int a = 0; a < 2; ++a)
        #pragma unroll
        for (int b = 0; b < 4; ++b)
            acc[a][b] = (v8f){0.f,0.f,0.f,0.f,0.f,0.f,0.f,0.f};

    // ---- pipeline prologue: tile 0 in flight
    issueA(0, 0);
    stageB(0, 0);

    for (int i = 0; i < nt; ++i) {
        const int cur = i & 1, nxt = cur ^ 1;

        if (i + 1 < nt) {
            issueA((i + 1) * 32, nxt);      // DMA next A panel (overlaps compute)
            stageB((i + 1) * 32, nxt);      // stage next B panel
            if (i + 2 < nt) {               // prefetch B panel two tiles ahead
                const unsigned short* pf =
                    B + (size_t)((i + 2) * 32 + (t >> 3)) * N + n_blk + (t & 7) * 16;
                __builtin_prefetch(pf, 0, 1);
            }
        }
#if HAS_TDM
        if (wave == 0) {                    // current DMA done; newest may remain
            if (i + 1 < nt) __builtin_amdgcn_s_wait_tensorcnt(1);
            else            __builtin_amdgcn_s_wait_tensorcnt(0);
        }
#endif
        __syncthreads();

        // ---- fragments from current buffers
        Frag af[2], bfm[4];
        #pragma unroll
        for (int mt = 0; mt < 2; ++mt) {
            int row = wm * 32 + mt * 16 + l16;            // A: interleaved K halves
            af[mt].i4[0] = *(const int4*)&As[cur][row][hf * 8];
            af[mt].i4[1] = *(const int4*)&As[cur][row][16 + hf * 8];
        }
        #pragma unroll
        for (int ntl = 0; ntl < 4; ++ntl) {
            int col = wn * 64 + ntl * 16 + l16;           // B: interleaved K halves
            bfm[ntl].i4[0] = *(const int4*)&Bs[cur][col][hf * 8];
            bfm[ntl].i4[1] = *(const int4*)&Bs[cur][col][16 + hf * 8];
        }
        #pragma unroll
        for (int mt = 0; mt < 2; ++mt)
            #pragma unroll
            for (int ntl = 0; ntl < 4; ++ntl)
                acc[mt][ntl] = __builtin_amdgcn_wmma_f32_16x16x32_bf16(
                    false, af[mt].v, false, bfm[ntl].v, (short)0, acc[mt][ntl],
                    false, false);

        __syncthreads();   // buffers free for next issue
    }

    // ---- epilogue
    #pragma unroll
    for (int mt = 0; mt < 2; ++mt)
        #pragma unroll
        for (int ntl = 0; ntl < 4; ++ntl)
            #pragma unroll
            for (int r = 0; r < 8; ++r) {
                int row = m_blk + wm * 32 + mt * 16 + r + 8 * hf;
                int col = n_blk + wn * 64 + ntl * 16 + l16;
                if (F32OUT)
                    ((float*)Cv)[(size_t)row * N + col] = acc[mt][ntl][r];
                else
                    ((unsigned short*)Cv)[(size_t)row * N + col] = f2bf(acc[mt][ntl][r]);
            }
}

// ---------------------------------------------------------------------------
// Kernel 3: RoPE in-place on q and k inside qkv buffer (bf16)
// ---------------------------------------------------------------------------
__global__ __launch_bounds__(256) void rope_kernel(unsigned short* __restrict__ qkv) {
    int idx = blockIdx.x * 256 + threadIdx.x;       // 2048 * 32 * 64
    int s   = idx >> 11;
    int rem = idx & 2047;
    int h   = rem >> 6;
    int d   = rem & 63;
    if (s >= SEQ) return;

    float freq = __expf(-(float)(2 * d) * (1.0f / (float)HD) * 9.210340371976184f);
    float ang  = (float)s * freq;
    float c, sn;
    __sincosf(ang, &sn, &c);

    size_t base = (size_t)s * QKV_N + h * HD + d;
    #pragma unroll
    for (int which = 0; which < 2; ++which) {        // 0 = q, 1 = k
        size_t o  = base + (size_t)which * HID;
        float x1  = __bfloat162float(__hip_bfloat16_raw{qkv[o]});
        float x2  = __bfloat162float(__hip_bfloat16_raw{qkv[o + 64]});
        qkv[o]      = f2bf(x1 * c - x2 * sn);
        qkv[o + 64] = f2bf(x2 * c + x1 * sn);
    }
}

// ---------------------------------------------------------------------------
// Kernel 4: flash attention, causal.
//   grid = (16 q-tiles, 32 heads); block = 256 (8 waves x 16 q rows).
// ---------------------------------------------------------------------------
__global__ __launch_bounds__(256) void attn_kernel(const unsigned short* __restrict__ qkv,
                                                   unsigned short* __restrict__ ctx) {
    __shared__ __align__(16) unsigned short Ks[32][136];     // [kv][d]
    __shared__ __align__(16) unsigned short Vt[128][40];     // [d][kv]
    __shared__ __align__(16) unsigned short Ps[8][16][40];   // per-wave P 16x32

    const int t    = threadIdx.x;
    const int wave = t >> 5;
    const int lane = t & 31;
    const int l16  = lane & 15;
    const int hf   = lane >> 4;
    const int qt   = blockIdx.x;
    const int h    = blockIdx.y;
    const int q0   = qt * 128 + wave * 16;
    const float scale = 0.08838834764831845f;  // 1/sqrt(128)

    const unsigned short* qp = qkv + (size_t)0 * HID + h * HD;
    const unsigned short* kp = qkv + (size_t)1 * HID + h * HD;
    const unsigned short* vp = qkv + (size_t)2 * HID + h * HD;

    Frag qf[4];
    {
        const unsigned short* qrow = qp + (size_t)(q0 + l16) * QKV_N;
        #pragma unroll
        for (int db = 0; db < 4; ++db) {
            qf[db].i4[0] = *(const int4*)(qrow + db * 32 + hf * 8);
            qf[db].i4[1] = *(const int4*)(qrow + db * 32 + 16 + hf * 8);
        }
    }

    v8f oacc[8];
    #pragma unroll
    for (int dtile = 0; dtile < 8; ++dtile)
        oacc[dtile] = (v8f){0.f,0.f,0.f,0.f,0.f,0.f,0.f,0.f};
    float mprev[8], lsum[8];
    #pragma unroll
    for (int r = 0; r < 8; ++r) { mprev[r] = -1e30f; lsum[r] = 0.f; }

    const int nkv = (qt + 1) * 4;
    for (int step = 0; step < nkv; ++step) {
        const int kv0 = step * 32;
        __syncthreads();
        // ---- stage K tile [32 kv][128 d]
        #pragma unroll
        for (int c = t; c < 512; c += 256) {
            int kv = c >> 4, off = c & 15;
            *(int4*)&Ks[kv][off * 8] =
                *(const int4*)(kp + (size_t)(kv0 + kv) * QKV_N + off * 8);
        }
        // ---- stage V^T tile [128 d][32 kv]
        {
            int dcol = t & 127, kh = t >> 7;
            union { unsigned short u[16]; int4 i4[2]; } tv;
            const unsigned short* vcol = vp + dcol;
            #pragma unroll
            for (int j = 0; j < 16; ++j)
                tv.u[j] = vcol[(size_t)(kv0 + kh * 16 + j) * QKV_N];
            *(int4*)&Vt[dcol][kh * 16]     = tv.i4[0];
            *(int4*)&Vt[dcol][kh * 16 + 8] = tv.i4[1];
        }
        // ---- prefetch next K/V tiles
        if (step + 1 < nkv) {
            size_t nrow = (size_t)(kv0 + 32 + (t >> 3)) * QKV_N + (t & 7) * 16;
            __builtin_prefetch(kp + nrow, 0, 1);
            __builtin_prefetch(vp + nrow, 0, 1);
        }
        __syncthreads();

        // ---- S = Q K^T for two 16-wide kv sub-tiles
        v8f s0 = (v8f){0.f,0.f,0.f,0.f,0.f,0.f,0.f,0.f};
        v8f s1 = s0;
        #pragma unroll
        for (int db = 0; db < 4; ++db) {
            Frag kf0, kf1;
            kf0.i4[0] = *(const int4*)&Ks[l16][db * 32 + hf * 8];
            kf0.i4[1] = *(const int4*)&Ks[l16][db * 32 + 16 + hf * 8];
            kf1.i4[0] = *(const int4*)&Ks[16 + l16][db * 32 + hf * 8];
            kf1.i4[1] = *(const int4*)&Ks[16 + l16][db * 32 + 16 + hf * 8];
            s0 = __builtin_amdgcn_wmma_f32_16x16x32_bf16(false, qf[db].v, false,
                     kf0.v, (short)0, s0, false, false);
            s1 = __builtin_amdgcn_wmma_f32_16x16x32_bf16(false, qf[db].v, false,
                     kf1.v, (short)0, s1, false, false);
        }

        // ---- online softmax
        #pragma unroll
        for (int r = 0; r < 8; ++r) {
            int qrow = q0 + r + 8 * hf;
            int c0 = kv0 + l16, c1 = c0 + 16;
            float a = s0[r] * scale + ((c0 <= qrow) ? 0.f : -1e30f);
            float b = s1[r] * scale + ((c1 <= qrow) ? 0.f : -1e30f);
            float rm   = rowmax16(fmaxf(a, b));
            float mn   = fmaxf(mprev[r], rm);
            float corr = __expf(mprev[r] - mn);
            float p0   = __expf(a - mn);
            float p1   = __expf(b - mn);
            float rs   = rowsum16(p0 + p1);
            lsum[r]  = lsum[r] * corr + rs;
            mprev[r] = mn;
            #pragma unroll
            for (int dtile = 0; dtile < 8; ++dtile) oacc[dtile][r] *= corr;
            int prow = r + 8 * hf;
            Ps[wave][prow][l16]      = f2bf(p0);
            Ps[wave][prow][16 + l16] = f2bf(p1);
        }
        __syncthreads();

        // ---- O += P V
        Frag pf;
        pf.i4[0] = *(const int4*)&Ps[wave][l16][hf * 8];
        pf.i4[1] = *(const int4*)&Ps[wave][l16][16 + hf * 8];
        #pragma unroll
        for (int dtile = 0; dtile < 8; ++dtile) {
            Frag vf;
            int d = dtile * 16 + l16;
            vf.i4[0] = *(const int4*)&Vt[d][hf * 8];
            vf.i4[1] = *(const int4*)&Vt[d][16 + hf * 8];
            oacc[dtile] = __builtin_amdgcn_wmma_f32_16x16x32_bf16(false, pf.v, false,
                              vf.v, (short)0, oacc[dtile], false, false);
        }
    }

    // ---- normalize + write context [s][h*128+d] as bf16
    #pragma unroll
    for (int dtile = 0; dtile < 8; ++dtile)
        #pragma unroll
        for (int r = 0; r < 8; ++r) {
            int srow = q0 + r + 8 * hf;
            int col  = h * HD + dtile * 16 + l16;
            float v  = oacc[dtile][r] * (1.0f / lsum[r]);
            ctx[(size_t)srow * HID + col] = f2bf(v);
        }
}

// ---------------------------------------------------------------------------
// Host-side launcher
// ---------------------------------------------------------------------------
extern "C" void kernel_launch(void* const* d_in, const int* in_sizes, int n_in,
                              void* d_out, int out_size, void* d_ws, size_t ws_size,
                              hipStream_t stream) {
    (void)in_sizes; (void)n_in; (void)out_size; (void)ws_size;
    const float* hs   = (const float*)d_in[0];   // [2048,4096]
    const float* qkvw = (const float*)d_in[1];   // [4096,12288]
    const float* ow   = (const float*)d_in[2];   // [4096,4096]
    // d_in[3] attention_mask: causal, computed analytically
    // d_in[4] position_ids:   arange, computed analytically

    unsigned short* hid_bf  = (unsigned short*)d_ws;
    unsigned short* qkvw_bf = hid_bf  + (size_t)SEQ * HID;
    unsigned short* ow_bf   = qkvw_bf + (size_t)HID * QKV_N;
    unsigned short* qkv     = ow_bf   + (size_t)HID * HID;
    unsigned short* ctx     = qkv     + (size_t)SEQ * QKV_N;

    cvt_f32_bf16<<<(SEQ * HID) / 1024, 256, 0, stream>>>(hs, hid_bf, (long long)SEQ * HID);
    cvt_f32_bf16<<<(HID * QKV_N) / 1024, 256, 0, stream>>>(qkvw, qkvw_bf, (long long)HID * QKV_N);
    cvt_f32_bf16<<<(HID * HID) / 1024, 256, 0, stream>>>(ow, ow_bf, (long long)HID * HID);

    gemm_bf16<false><<<dim3(QKV_N / 128, SEQ / 128), 256, 0, stream>>>(
        hid_bf, qkvw_bf, qkv, SEQ, QKV_N, HID);

    rope_kernel<<<(SEQ * NH * (HD / 2)) / 256, 256, 0, stream>>>(qkv);

    attn_kernel<<<dim3(SEQ / 128, NH), 256, 0, stream>>>(qkv, ctx);

    gemm_bf16<true><<<dim3(HID / 128, SEQ / 128), 256, 0, stream>>>(
        ctx, ow_bf, (float*)d_out, SEQ, HID, HID);
}
